// InferenceVQVAE_43404939493735
// MI455X (gfx1250) — matrix operand order repaired and checked by reading, not compiled, not run on hardware
//
#include <hip/hip_runtime.h>
#include <stdint.h>

// ---------------------------------------------------------------------------
// CDNA5 (gfx1250) wave32 WMMA types
// ---------------------------------------------------------------------------
typedef __bf16        v16bf __attribute__((ext_vector_type(16)));
typedef float         v8f   __attribute__((ext_vector_type(8)));
typedef unsigned int  u32x4 __attribute__((ext_vector_type(4)));

union Frag { u32x4 q[2]; v16bf v; };

__device__ __forceinline__ unsigned short f2bf(float f) {
  union { float f; unsigned u; } v; v.f = f;
  unsigned r = v.u + 0x7FFFu + ((v.u >> 16) & 1u);   // round-nearest-even
  return (unsigned short)(r >> 16);
}
__device__ __forceinline__ float bf2f(unsigned short h) {
  union { unsigned u; float f; } v; v.u = ((unsigned)h) << 16;
  return v.f;
}

// ---------------------------------------------------------------------------
// Async global->LDS staging (GLOBAL_LOAD_ASYNC_TO_LDS_B128, ASYNCcnt tracked).
// Builtin signature (from hipcc diagnostic): (v4i AS1*, v4i AS3*, imm, imm).
// AS casts via integer round-trip: generic LDS ptr low 32 bits == AS3 offset.
// ---------------------------------------------------------------------------
#if defined(__gfx1250__) && __has_builtin(__builtin_amdgcn_global_load_async_to_lds_b128)
#define HAVE_ASYNC_LDS 1
#else
#define HAVE_ASYNC_LDS 0
#endif

typedef int v4i_vs __attribute__((vector_size(16)));
typedef __attribute__((address_space(1))) v4i_vs* as1_v4i;
typedef __attribute__((address_space(3))) v4i_vs* as3_v4i;

__device__ __forceinline__ void async_copy_b128(const unsigned int* g, unsigned int* l) {
#if HAVE_ASYNC_LDS
  __builtin_amdgcn_global_load_async_to_lds_b128(
      (as1_v4i)(uintptr_t)(g),
      (as3_v4i)(unsigned int)(uintptr_t)(l), 0, 0);
#else
  (void)g; (void)l;
#endif
}

__device__ __forceinline__ void wait_async0() {
#if HAVE_ASYNC_LDS
#if __has_builtin(__builtin_amdgcn_s_wait_asynccnt)
  __builtin_amdgcn_s_wait_asynccnt(0);
#else
  asm volatile("s_wait_asynccnt 0x0" ::: "memory");
#endif
#endif
}

// ---------------------------------------------------------------------------
// Weight / codebook repack kernels (tiny, memory-bound)
// ---------------------------------------------------------------------------
__global__ __launch_bounds__(256) void k_pack_conv2_w(const float* __restrict__ w2,
                                                      unsigned short* __restrict__ A2) {
  // A2[e][k], k = c*64 + tap ; enc_w2 layout [E=256][H=128][4,4,4]
  unsigned gid = blockIdx.x * 256u + threadIdx.x;      // 2,097,152
  unsigned e = gid >> 13, k = gid & 8191u;
  A2[gid] = f2bf(w2[(size_t)((e << 7) + (k >> 6)) * 64u + (k & 63u)]);
}

__global__ __launch_bounds__(256) void k_pack_convt1_w(const float* __restrict__ wd1,
                                                       unsigned short* __restrict__ A1) {
  // 8 parity classes: A1[cls][c][k], k = t*256 + e; dec_w1 layout [E=256][H=128][4,4,4]
  unsigned gid = blockIdx.x * 256u + threadIdx.x;      // 2,097,152
  unsigned cls = gid >> 18;
  unsigned rem = gid & 262143u;
  unsigned c = rem >> 11, k = rem & 2047u;
  unsigned t = k >> 8, e = k & 255u;
  int pd = (cls >> 2) & 1, ph = (cls >> 1) & 1, pw = cls & 1;
  int td = (t >> 2) & 1,  th = (t >> 1) & 1,  tw = t & 1;
  int kd = (1 - pd) + 2 * td, kh = (1 - ph) + 2 * th, kw = (1 - pw) + 2 * tw;
  A1[gid] = f2bf(wd1[(size_t)((e << 7) + c) * 64u + (unsigned)(kd * 16 + kh * 4 + kw)]);
}

__global__ __launch_bounds__(256) void k_pack_cb(const float* __restrict__ cb,
                                                 unsigned short* __restrict__ cbb) {
  unsigned gid = blockIdx.x * 256u + threadIdx.x;      // 262,144
  cbb[gid] = f2bf(cb[gid]);
}

__global__ __launch_bounds__(256) void k_cnorm(const float* __restrict__ cb,
                                               float* __restrict__ cn) {
  unsigned i = blockIdx.x * 256u + threadIdx.x;        // 1024
  const float* row = cb + (size_t)i * 256u;
  float s = 0.f;
  for (int j = 0; j < 256; ++j) s += row[j] * row[j];
  cn[i] = s;
}

// ---------------------------------------------------------------------------
// Encoder conv1: 1 -> 128 ch, k4 s2 p1, 16^3 -> 8^3, ReLU, bf16 out (Cin=1 -> VALU)
// ---------------------------------------------------------------------------
__global__ __launch_bounds__(256) void k_conv1(const float* __restrict__ x,
                                               const float* __restrict__ w1,
                                               const float* __restrict__ b1,
                                               unsigned short* __restrict__ hb) {
  unsigned gid = blockIdx.x * 256u + threadIdx.x;      // 67,108,864 = (b*128+c)*512+vox
  unsigned b   = gid >> 16;
  unsigned c   = (gid >> 9) & 127u;
  unsigned vox = gid & 511u;
  int oz = vox >> 6, oy = (vox >> 3) & 7, ox = vox & 7;
  const float* xb = x  + (size_t)b * 4096u;
  const float* wc = w1 + (size_t)c * 64u;
  float acc = b1[c];
#pragma unroll
  for (int kd = 0; kd < 4; ++kd) {
    int iz = 2 * oz - 1 + kd;
    if ((unsigned)iz >= 16u) continue;
#pragma unroll
    for (int kh = 0; kh < 4; ++kh) {
      int iy = 2 * oy - 1 + kh;
      if ((unsigned)iy >= 16u) continue;
#pragma unroll
      for (int kw = 0; kw < 4; ++kw) {
        int ix = 2 * ox - 1 + kw;
        if ((unsigned)ix >= 16u) continue;
        acc += xb[iz * 256 + iy * 16 + ix] * wc[kd * 16 + kh * 4 + kw];
      }
    }
  }
  hb[gid] = f2bf(fmaxf(acc, 0.0f));
}

// ---------------------------------------------------------------------------
// Encoder conv2 as implicit GEMM: C[256][65536] = A[256][8192] * B(im2col of h)
// Block: 256 thr (8 waves), tile 128(M) x 64(N), K-chunk 32, WMMA bf16 16x16x32
// A tile staged via async global->LDS (overlapped with the im2col B gather).
// ---------------------------------------------------------------------------
__global__ __launch_bounds__(256) void k_conv2_gemm(
    const unsigned short* __restrict__ A2,   // [256][8192] bf16
    const unsigned short* __restrict__ hb,   // [1024][128][512] bf16
    const float* __restrict__ b2,            // [256]
    unsigned short* __restrict__ zb) {       // [65536][256] bf16
  __shared__ __attribute__((aligned(16))) unsigned int sA[128 * 16]; // 128 rows x 32 bf16
  __shared__ __attribute__((aligned(16))) unsigned int sB[64 * 16];  // 64 cols  x 32 bf16
  const unsigned tid = threadIdx.x, lane = tid & 31u, wid = tid >> 5;
  const unsigned bm = blockIdx.x & 1u, bn = blockIdx.x >> 1;
  const unsigned m0 = bm << 7, n0 = bn << 6;
  const unsigned wm = (wid & 3u) << 5, wn = (wid >> 2) << 5;  // 4x2 wave grid
  const unsigned l15 = lane & 15u, hi = lane >> 4;
  const unsigned ac0 = hi << 3;   // A fragment dword base: 0 / 8
  const unsigned bc0 = hi << 3;   // B fragment dword base: 0 / 8
  const unsigned int* A2u = (const unsigned int*)A2;

  v8f acc[2][2];
#pragma unroll
  for (int i = 0; i < 2; ++i)
#pragma unroll
    for (int j = 0; j < 2; ++j) {
      v8f z = {0.f, 0.f, 0.f, 0.f, 0.f, 0.f, 0.f, 0.f};
      acc[i][j] = z;
    }

  for (unsigned k0 = 0; k0 < 8192u; k0 += 32u) {
    // --- stage A tile: 128 x 16 dwords (8 dwords per thread) ---
#if HAVE_ASYNC_LDS
    {
      unsigned base = tid << 3;                 // 0..2047, 8 contiguous dwords
      unsigned m = base >> 4, d = base & 15u;
      const unsigned int* g = A2u + (size_t)(m0 + m) * 4096u + (k0 >> 1) + d;
      async_copy_b128(g,     &sA[base]);
      async_copy_b128(g + 4, &sA[base + 4]);
    }
#else
#pragma unroll
    for (unsigned i = 0; i < 8; ++i) {
      unsigned idx = tid + (i << 8);
      unsigned m = idx >> 4, d = idx & 15u;
      sA[idx] = A2u[(size_t)(m0 + m) * 4096u + (k0 >> 1) + d];
    }
#endif
    if (k0 + 32u < 8192u)
      __builtin_prefetch(A2u + (size_t)(m0 + (tid & 127u)) * 4096u + ((k0 + 32u) >> 1), 0, 1);
    // --- im2col gather B tile: 64 x 16 dwords (k = c*64 + tap) ---
    unsigned ch = k0 >> 6, tap0 = k0 & 32u;
#pragma unroll
    for (unsigned i = 0; i < 4; ++i) {
      unsigned idx = tid + (i << 8);
      unsigned nl = idx >> 4, dd = idx & 15u;
      unsigned n = n0 + nl;
      unsigned b = n >> 6, vox = n & 63u;
      int oz = vox >> 4, oy = (vox >> 2) & 3, ox = vox & 3;
      unsigned val = 0;
#pragma unroll
      for (int j = 0; j < 2; ++j) {
        unsigned tap = tap0 + (dd << 1) + (unsigned)j;
        int kd = tap >> 4, kh = (tap >> 2) & 3, kw = tap & 3;
        int iz = 2 * oz - 1 + kd, iy = 2 * oy - 1 + kh, ix = 2 * ox - 1 + kw;
        unsigned short hv = 0;
        if ((unsigned)iz < 8u && (unsigned)iy < 8u && (unsigned)ix < 8u)
          hv = hb[(size_t)((b << 7) + ch) * 512u + (unsigned)(iz * 64 + iy * 8 + ix)];
        val |= ((unsigned)hv) << (16 * j);
      }
      sB[idx] = val;
    }
    wait_async0();
    __syncthreads();
    // --- fragments + 2x2 WMMA ---
    Frag aF[2], bF[2];
#pragma unroll
    for (int mi = 0; mi < 2; ++mi) {
      const unsigned int* p = &sA[(wm + ((unsigned)mi << 4) + l15) * 16u + ac0];
      aF[mi].q[0] = *(const u32x4*)(p);
      aF[mi].q[1] = *(const u32x4*)(p + 4);
    }
#pragma unroll
    for (int ni = 0; ni < 2; ++ni) {
      const unsigned int* p = &sB[(wn + ((unsigned)ni << 4) + l15) * 16u + bc0];
      bF[ni].q[0] = *(const u32x4*)(p);
      bF[ni].q[1] = *(const u32x4*)(p + 4);
    }
#pragma unroll
    for (int mi = 0; mi < 2; ++mi)
#pragma unroll
      for (int ni = 0; ni < 2; ++ni)
        acc[mi][ni] = __builtin_amdgcn_wmma_f32_16x16x32_bf16(
            false, aF[mi].v, false, bF[ni].v, (short)0, acc[mi][ni], false, false);
    __syncthreads();
  }
  // --- store C (bias fused): z[n][m] bf16 ---
#pragma unroll
  for (int mi = 0; mi < 2; ++mi)
#pragma unroll
    for (int ni = 0; ni < 2; ++ni)
#pragma unroll
      for (int r = 0; r < 8; ++r) {
        unsigned m = m0 + wm + ((unsigned)mi << 4) + (unsigned)r + (hi << 3);
        unsigned n = n0 + wn + ((unsigned)ni << 4) + l15;
        zb[(size_t)n * 256u + m] = f2bf(acc[mi][ni][r] + b2[m]);
      }
}

// ---------------------------------------------------------------------------
// VQ: per-row argmin over 1024 codes of (||c||^2 - 2 z.c) via WMMA GEMM.
// Block 256 thr = 8 waves; each wave owns 16 rows; 64 code tiles of 16.
// ---------------------------------------------------------------------------
__global__ __launch_bounds__(256) void k_vq_argmin(
    const unsigned short* __restrict__ zb,   // [65536][256] bf16
    const unsigned short* __restrict__ cbb,  // [1024][256]  bf16
    const float* __restrict__ cn,            // [1024]
    int* __restrict__ outIdx) {              // [65536]
  __shared__ __attribute__((aligned(16))) unsigned int sC[16 * 128]; // 16 codes x 256 bf16
  __shared__ float sD[8][16][17];
  const unsigned tid = threadIdx.x, lane = tid & 31u, wid = tid >> 5;
  const unsigned l15 = lane & 15u, hi = lane >> 4;
  const unsigned ac0 = hi << 3, bc0 = hi << 3;
  const unsigned nrow = blockIdx.x * 128u + (wid << 4) + l15;
  const unsigned int* zu = (const unsigned int*)zb;
  const unsigned int* cu = (const unsigned int*)cbb;

  // Keep the whole K=256 A (z-rows) fragment set in registers (8 k-steps).
  Frag aF[8];
#pragma unroll
  for (int ks = 0; ks < 8; ++ks) {
    const unsigned int* p = zu + (size_t)nrow * 128u + ((unsigned)ks << 4) + ac0;
    aF[ks].q[0] = *(const u32x4*)(p);
    aF[ks].q[1] = *(const u32x4*)(p + 4);
  }

  float minv = 3.4e38f;
  int   mini = 0;
  for (unsigned ct = 0; ct < 64u; ++ct) {
    // stage 16-code tile of the codebook
#pragma unroll
    for (unsigned i = 0; i < 8; ++i)
      sC[tid + (i << 8)] = cu[(size_t)ct * 2048u + tid + (i << 8)];
    __syncthreads();

    v8f acc = {0.f, 0.f, 0.f, 0.f, 0.f, 0.f, 0.f, 0.f};
#pragma unroll
    for (int ks = 0; ks < 8; ++ks) {
      Frag bF;
      const unsigned int* p = &sC[l15 * 128u + ((unsigned)ks << 4) + bc0];
      bF.q[0] = *(const u32x4*)(p);
      bF.q[1] = *(const u32x4*)(p + 4);
      acc = __builtin_amdgcn_wmma_f32_16x16x32_bf16(
          false, aF[ks].v, false, bF.v, (short)0, acc, false, false);
    }
    unsigned code = (ct << 4) + l15;
    float cnv = cn[code];
#pragma unroll
    for (int r = 0; r < 8; ++r)
      sD[wid][(unsigned)r + (hi << 3)][l15] = cnv - 2.0f * acc[r];
    // same-wave LDS is in-order: lanes 0..15 scan their row across 16 codes
    if (hi == 0) {
#pragma unroll
      for (int cc = 0; cc < 16; ++cc) {
        float v = sD[wid][l15][cc];
        if (v < minv) { minv = v; mini = (int)((ct << 4) + (unsigned)cc); }
      }
    }
    __syncthreads();
  }
  if (hi == 0) outIdx[blockIdx.x * 128u + (wid << 4) + l15] = mini;
}

// ---------------------------------------------------------------------------
// Quantize gather: q[n][e] = codebook_bf16[idx[n]][e] (dword granularity)
// ---------------------------------------------------------------------------
__global__ __launch_bounds__(256) void k_qgather(const int* __restrict__ idx,
                                                 const unsigned short* __restrict__ cbb,
                                                 unsigned short* __restrict__ q) {
  unsigned gid = blockIdx.x * 256u + threadIdx.x;    // 8,388,608 dwords
  unsigned n = gid >> 7, d = gid & 127u;
  ((unsigned int*)q)[gid] = ((const unsigned int*)cbb)[(size_t)idx[n] * 128u + d];
}

// ---------------------------------------------------------------------------
// Decoder convT1 (256->128, k4 s2 p1, 4^3 -> 8^3) as 8 parity-class GEMMs:
// per class: C[128][65536] = A1[cls][128][2048] * B(gather of q), ReLU+bias fused.
// ---------------------------------------------------------------------------
__global__ __launch_bounds__(256) void k_convt1_gemm(
    const unsigned short* __restrict__ A1,   // [8][128][2048] bf16
    const unsigned short* __restrict__ q,    // [65536][256]   bf16
    const float* __restrict__ bd1,           // [128]
    unsigned short* __restrict__ h2) {       // [1024][512][128] bf16
  __shared__ __attribute__((aligned(16))) unsigned int sA[128 * 16];
  __shared__ __attribute__((aligned(16))) unsigned int sB[64 * 16];
  const unsigned tid = threadIdx.x, lane = tid & 31u, wid = tid >> 5;
  const unsigned cls = blockIdx.x >> 10;     // 8 parity classes
  const unsigned bn  = blockIdx.x & 1023u;
  const unsigned n0  = bn << 6;
  const unsigned wm = (wid & 3u) << 5, wn = (wid >> 2) << 5;
  const unsigned l15 = lane & 15u, hi = lane >> 4;
  const unsigned ac0 = hi << 3, bc0 = hi << 3;
  const int pd = (cls >> 2) & 1, ph = (cls >> 1) & 1, pw = cls & 1;
  const unsigned int* Au = (const unsigned int*)A1 + (size_t)cls * (128u * 1024u);
  const unsigned int* qu = (const unsigned int*)q;

  v8f acc[2][2];
#pragma unroll
  for (int i = 0; i < 2; ++i)
#pragma unroll
    for (int j = 0; j < 2; ++j) {
      v8f z = {0.f, 0.f, 0.f, 0.f, 0.f, 0.f, 0.f, 0.f};
      acc[i][j] = z;
    }

  for (unsigned k0 = 0; k0 < 2048u; k0 += 32u) {
#if HAVE_ASYNC_LDS
    {
      unsigned base = tid << 3;
      unsigned m = base >> 4, d = base & 15u;
      const unsigned int* g = Au + (size_t)m * 1024u + (k0 >> 1) + d;
      async_copy_b128(g,     &sA[base]);
      async_copy_b128(g + 4, &sA[base + 4]);
    }
#else
#pragma unroll
    for (unsigned i = 0; i < 8; ++i) {
      unsigned idx = tid + (i << 8);
      unsigned m = idx >> 4, d = idx & 15u;
      sA[idx] = Au[(size_t)m * 1024u + (k0 >> 1) + d];
    }
#endif
    unsigned t = k0 >> 8, e0 = k0 & 255u;    // k = t*256 + e
    int td = (t >> 2) & 1, th = (t >> 1) & 1, tw = t & 1;
#pragma unroll
    for (unsigned i = 0; i < 4; ++i) {
      unsigned idx = tid + (i << 8);
      unsigned nl = idx >> 4, dd = idx & 15u;
      unsigned n = n0 + nl;
      unsigned b = n >> 6, vox = n & 63u;
      int ozc = vox >> 4, oyc = (vox >> 2) & 3, oxc = vox & 3;
      int iz = ozc + pd - td, iy = oyc + ph - th, ix = oxc + pw - tw;
      unsigned val = 0;
      if ((unsigned)iz < 4u && (unsigned)iy < 4u && (unsigned)ix < 4u)
        val = qu[(size_t)((b << 6) + (unsigned)(iz * 16 + iy * 4 + ix)) * 128u + (e0 >> 1) + dd];
      sB[idx] = val;
    }
    wait_async0();
    __syncthreads();
    Frag aF[2], bF[2];
#pragma unroll
    for (int mi = 0; mi < 2; ++mi) {
      const unsigned int* p = &sA[(wm + ((unsigned)mi << 4) + l15) * 16u + ac0];
      aF[mi].q[0] = *(const u32x4*)(p);
      aF[mi].q[1] = *(const u32x4*)(p + 4);
    }
#pragma unroll
    for (int ni = 0; ni < 2; ++ni) {
      const unsigned int* p = &sB[(wn + ((unsigned)ni << 4) + l15) * 16u + bc0];
      bF[ni].q[0] = *(const u32x4*)(p);
      bF[ni].q[1] = *(const u32x4*)(p + 4);
    }
#pragma unroll
    for (int mi = 0; mi < 2; ++mi)
#pragma unroll
      for (int ni = 0; ni < 2; ++ni)
        acc[mi][ni] = __builtin_amdgcn_wmma_f32_16x16x32_bf16(
            false, aF[mi].v, false, bF[ni].v, (short)0, acc[mi][ni], false, false);
    __syncthreads();
  }
  // store: h2[(b*512 + ov)*128 + c] = relu(acc + bias)
#pragma unroll
  for (int mi = 0; mi < 2; ++mi)
#pragma unroll
    for (int ni = 0; ni < 2; ++ni)
#pragma unroll
      for (int r = 0; r < 8; ++r) {
        unsigned c = wm + ((unsigned)mi << 4) + (unsigned)r + (hi << 3);
        unsigned n = n0 + wn + ((unsigned)ni << 4) + l15;
        unsigned b = n >> 6, vox = n & 63u;
        int ozc = vox >> 4, oyc = (vox >> 2) & 3, oxc = vox & 3;
        unsigned ov = (unsigned)((2 * ozc + pd) * 64 + (2 * oyc + ph) * 8 + (2 * oxc + pw));
        float v = fmaxf(acc[mi][ni][r] + bd1[c], 0.0f);
        h2[(size_t)((b << 9) + ov) * 128u + c] = f2bf(v);
      }
}

// ---------------------------------------------------------------------------
// Decoder convT2 (128 -> 1 ch, 8^3 -> 16^3): Cout=1 -> VALU dot products, f32 out
// ---------------------------------------------------------------------------
__global__ __launch_bounds__(256) void k_convt2(const unsigned short* __restrict__ h2,
                                                const float* __restrict__ wd2,  // [128][64]
                                                const float* __restrict__ bd2,  // [1]
                                                float* __restrict__ out) {
  unsigned gid = blockIdx.x * 256u + threadIdx.x;     // 4,194,304
  unsigned b = gid >> 12, ov = gid & 4095u;
  int od = ov >> 8, oh = (ov >> 4) & 15, ow = ov & 15;
  int pd = od & 1, ph = oh & 1, pw = ow & 1;
  float acc = bd2[0];
  for (int td = 0; td < 2; ++td) {
    int iz = (od >> 1) + pd - td;
    if ((unsigned)iz >= 8u) continue;
    int kd = (1 - pd) + 2 * td;
    for (int th = 0; th < 2; ++th) {
      int iy = (oh >> 1) + ph - th;
      if ((unsigned)iy >= 8u) continue;
      int kh = (1 - ph) + 2 * th;
      for (int tw = 0; tw < 2; ++tw) {
        int ix = (ow >> 1) + pw - tw;
        if ((unsigned)ix >= 8u) continue;
        int kw = (1 - pw) + 2 * tw;
        const unsigned short* hrow = h2 + (size_t)((b << 9) + (unsigned)(iz * 64 + iy * 8 + ix)) * 128u;
        const float* wt = wd2 + (kd * 16 + kh * 4 + kw);
        float s = 0.f;
#pragma unroll 8
        for (int c = 0; c < 128; ++c) s += bf2f(hrow[c]) * wt[c * 64];
        acc += s;
      }
    }
  }
  out[gid] = acc;
}

// ---------------------------------------------------------------------------
// Launch: encoder -> VQ -> decoder, all on `stream`, scratch carved from d_ws.
// ---------------------------------------------------------------------------
extern "C" void kernel_launch(void* const* d_in, const int* in_sizes, int n_in,
                              void* d_out, int out_size, void* d_ws, size_t ws_size,
                              hipStream_t stream) {
  (void)in_sizes; (void)n_in; (void)out_size; (void)ws_size;
  const float* x      = (const float*)d_in[0];
  const float* enc_w1 = (const float*)d_in[1];
  const float* enc_b1 = (const float*)d_in[2];
  const float* enc_w2 = (const float*)d_in[3];
  const float* enc_b2 = (const float*)d_in[4];
  const float* cbf    = (const float*)d_in[5];
  const float* dec_w1 = (const float*)d_in[6];
  const float* dec_b1 = (const float*)d_in[7];
  const float* dec_w2 = (const float*)d_in[8];
  const float* dec_b2 = (const float*)d_in[9];
  float* out = (float*)d_out;

  char* ws = (char*)d_ws;
  unsigned short* hb  = (unsigned short*)(ws + 0ull);           // 134,217,728 B
  unsigned short* zb  = (unsigned short*)(ws + 134217728ull);   //  33,554,432 B
  int*            vqi = (int*)           (ws + 167772160ull);   //     262,144 B
  unsigned short* q   = (unsigned short*)(ws + 168034304ull);   //  33,554,432 B
  unsigned short* h2  = (unsigned short*)(ws + 201588736ull);   // 134,217,728 B
  unsigned short* A2  = (unsigned short*)(ws + 335806464ull);   //   4,194,304 B
  unsigned short* A1  = (unsigned short*)(ws + 340000768ull);   //   4,194,304 B
  unsigned short* cbb = (unsigned short*)(ws + 344195072ull);   //     524,288 B
  float*          cn  = (float*)         (ws + 344719360ull);   //       4,096 B

  k_pack_conv2_w <<<8192,   256, 0, stream>>>(enc_w2, A2);
  k_pack_convt1_w<<<8192,   256, 0, stream>>>(dec_w1, A1);
  k_pack_cb      <<<1024,   256, 0, stream>>>(cbf, cbb);
  k_cnorm        <<<4,      256, 0, stream>>>(cbf, cn);
  k_conv1        <<<262144, 256, 0, stream>>>(x, enc_w1, enc_b1, hb);
  k_conv2_gemm   <<<2048,   256, 0, stream>>>(A2, hb, enc_b2, zb);
  k_vq_argmin    <<<512,    256, 0, stream>>>(zb, cbb, cn, vqi);
  k_qgather      <<<32768,  256, 0, stream>>>(vqi, cbb, q);
  k_convt1_gemm  <<<8192,   256, 0, stream>>>(A1, q, dec_b1, h2);
  k_convt2       <<<16384,  256, 0, stream>>>(h2, dec_w2, dec_b2, out);
}
